// Gaitnet_10101763080312
// MI455X (gfx1250) — compile-verified, matrix operand order checked by reading, not run on hardware
//
#include <hip/hip_runtime.h>
#include <hip/hip_bf16.h>

typedef __attribute__((ext_vector_type(16))) _Float16 v16h;
typedef __attribute__((ext_vector_type(8)))  float    v8f;

// ---------------- problem constants ----------------
static constexpr int B_    = 8192;   // batch
static constexpr int NOPT  = 64;     // footstep options
static constexpr int OPTS  = 65;     // options + no-act row
// workspace layout (in _Float16 elements). total ~2.3 MB.
static constexpr int OFF_RW1P = 0;        // 128 x 32
static constexpr int OFF_RW2P = 4096;     // 128 x 128
static constexpr int OFF_FW1P = 20480;    // 64 x 32
static constexpr int OFF_FW2P = 22528;    // 64 x 64
static constexpr int OFF_TW1P = 26624;    // 256 x 192
static constexpr int OFF_TW2P = 75776;    // 128 x 256
static constexpr int OFF_RFP  = 108544;   // 8192 x 128 (rf, A-fragment order)
static constexpr int PACK_TOTAL = 108544;

#define DEV __device__ __forceinline__

// Intra-wave LDS ordering: each wave owns a private LDS slice, so stage
// boundaries need only "my DS stores have landed before my cross-lane DS
// loads" -- that is s_wait_dscnt 0 on CDNA5 (DScnt tracks DS ops in-order).
// No workgroup barrier: the waves of a block free-run independently.
DEV void wave_lds_sync() {
  __builtin_amdgcn_wave_barrier();
  asm volatile("s_wait_dscnt 0" ::: "memory");
  __builtin_amdgcn_wave_barrier();
}

// ---------------- fragment helpers ----------------
// A-fragment (16xK f16 tile in LDS, row-major, stride in halves):
//   lane = (m & 15) + 16*k_half ; halves h -> k = ks*32 + k_half*16 + h
// B-fragment (pre-packed weights): tile (nt,ks) stored as 32 lanes x 16 halves.
// C layout (per ISA): lanes 0-15: VGPR v -> M=v ; lanes 16-31: VGPR v -> M=v+8; N=lane&15.
//
// Loop order is ks-outer / nt-inner with NT live accumulators: each A fragment
// is loaded from LDS exactly once per K-step and feeds NT consecutive WMMAs
// (the pattern the matrix pipe's operand caching favors).

template <int NT, int KS>
DEV void wave_gemm_lds(const _Float16* as, int astride, const _Float16* wp,
                       float* cs, int cstride, int lane, int a_col_off,
                       int col_base) {
  const int m  = lane & 15;
  const int kh = (lane >> 4) * 16;
  const int rh = (lane >> 4) * 8;
  const v8f zero = {};
  v8f acc[NT];
#pragma unroll
  for (int nt = 0; nt < NT; ++nt) acc[nt] = zero;
#pragma unroll
  for (int ks = 0; ks < KS; ++ks) {
    v16h a = *(const v16h*)(as + m * astride + a_col_off + ks * 32 + kh);
#pragma unroll
    for (int nt = 0; nt < NT; ++nt) {
      v16h b = *(const v16h*)(wp + (((nt * KS) + ks) * 32 + lane) * 16);
      acc[nt] = __builtin_amdgcn_wmma_f32_16x16x32_f16(false, a, false, b,
                                                       (short)0, acc[nt],
                                                       false, false);
    }
  }
#pragma unroll
  for (int nt = 0; nt < NT; ++nt) {
    const int col = col_base + nt * 16 + m;
#pragma unroll
    for (int v = 0; v < 8; ++v) cs[(rh + v) * cstride + col] = acc[nt][v];
  }
}

// trunk layer-1: A = [rf (cols 0..127, global fragment-packed) | ff (cols 128..191, LDS)]
template <int NT>
DEV void wave_gemm_trunk1(const _Float16* as, int astride,
                          const _Float16* rfp_tile, const _Float16* wp,
                          float* cs, int cstride, int lane, int col_base) {
  const int m  = lane & 15;
  const int kh = (lane >> 4) * 16;
  const int rh = (lane >> 4) * 8;
  const v8f zero = {};
  v8f acc[NT];
#pragma unroll
  for (int nt = 0; nt < NT; ++nt) acc[nt] = zero;
#pragma unroll
  for (int ks = 0; ks < 6; ++ks) {
    v16h a;
    if (ks < 4) a = *(const v16h*)(rfp_tile + ks * 512 + lane * 16);
    else        a = *(const v16h*)(as + m * astride + 128 + (ks - 4) * 32 + kh);
#pragma unroll
    for (int nt = 0; nt < NT; ++nt) {
      v16h b = *(const v16h*)(wp + ((nt * 6 + ks) * 32 + lane) * 16);
      acc[nt] = __builtin_amdgcn_wmma_f32_16x16x32_f16(false, a, false, b,
                                                       (short)0, acc[nt],
                                                       false, false);
    }
  }
#pragma unroll
  for (int nt = 0; nt < NT; ++nt) {
    const int col = col_base + nt * 16 + m;
#pragma unroll
    for (int v = 0; v < 8; ++v) cs[(rh + v) * cstride + col] = acc[nt][v];
  }
}

// LayerNorm(+bias) -> *gamma + beta -> ReLU -> f16 into LDS A-stage.
// Row r handled by lane pair (r, r+16); each half sums W/2 cols, combined via shfl_xor(16).
DEV void ln_relu_to_f16(const float* cs, int cstride, _Float16* as, int astride,
                        int W, int lane, const float* __restrict__ bias,
                        const float* __restrict__ g,
                        const float* __restrict__ be, int out_col_off) {
  const int r = lane & 15, h = lane >> 4;
  const int c0 = h * (W >> 1), c1 = c0 + (W >> 1);
  float sum = 0.f, ss = 0.f;
  for (int c = c0; c < c1; ++c) {
    float x = cs[r * cstride + c] + bias[c];
    sum += x; ss += x * x;
  }
  sum += __shfl_xor(sum, 16, 32);
  ss  += __shfl_xor(ss, 16, 32);
  const float mean = sum / (float)W;
  const float rstd = rsqrtf(ss / (float)W - mean * mean + 1e-5f);
  for (int c = c0; c < c1; ++c) {
    float x = cs[r * cstride + c] + bias[c];
    float y = fmaxf((x - mean) * rstd * g[c] + be[c], 0.f);
    as[r * astride + out_col_off + c] = (_Float16)y;
  }
}

// ---------------- kernel 0: pack all weights to f16 fragment order ----------------
__global__ void __launch_bounds__(256) gait_pack_kernel(
    const float* rw1, const float* rw2, const float* fw1, const float* fw2,
    const float* tw1, const float* tw2, _Float16* ws) {
  for (int idx = blockIdx.x * blockDim.x + threadIdx.x; idx < PACK_TOTAL;
       idx += gridDim.x * blockDim.x) {
    const float* W; _Float16* dst; int Kin, Kpad, p;
    if (idx < 4096)       { W = rw1; dst = ws + OFF_RW1P; Kin = 22;  Kpad = 32;  p = idx; }
    else if (idx < 20480) { W = rw2; dst = ws + OFF_RW2P; Kin = 128; Kpad = 128; p = idx - 4096; }
    else if (idx < 22528) { W = fw1; dst = ws + OFF_FW1P; Kin = 7;   Kpad = 32;  p = idx - 20480; }
    else if (idx < 26624) { W = fw2; dst = ws + OFF_FW2P; Kin = 64;  Kpad = 64;  p = idx - 22528; }
    else if (idx < 75776) { W = tw1; dst = ws + OFF_TW1P; Kin = 192; Kpad = 192; p = idx - 26624; }
    else                  { W = tw2; dst = ws + OFF_TW2P; Kin = 256; Kpad = 256; p = idx - 75776; }
    const int tile = p >> 9, w = p & 511;
    const int lane = w >> 4, hh = w & 15;
    const int ksteps = Kpad >> 5;
    const int nt = tile / ksteps, ks = tile - nt * ksteps;
    const int n = nt * 16 + (lane & 15);
    const int k = ks * 32 + (lane >> 4) * 16 + hh;
    // D[m,n] = sum_k A[m,k]*W[n,k]  ->  B[k,n] = W[n,k]
    const float v = (k < Kin) ? W[n * Kin + k] : 0.0f;
    dst[p] = (_Float16)v;
  }
}

// ---------------- kernel 1: robot MLP (22 -> 128 -> 128), rf in fragment order ----------------
__global__ void __launch_bounds__(128) gait_robot_kernel(
    const float* __restrict__ obs, const _Float16* __restrict__ wsH,
    _Float16* __restrict__ rfp,
    const float* rb1, const float* rg1, const float* rbe1,
    const float* rb2, const float* rg2, const float* rbe2) {
  __shared__ alignas(32) _Float16 aS[4][16 * 128];
  __shared__ alignas(32) float    cS[4][16 * 128];
  const int lane = threadIdx.x & 31, wave = threadIdx.x >> 5;
  const int bt = blockIdx.x * 4 + wave;           // 512 row tiles of 16
  _Float16* as = aS[wave];
  float*    cs = cS[wave];
  const _Float16* rw1p = wsH + OFF_RW1P;
  const _Float16* rw2p = wsH + OFF_RW2P;

  {  // stage robot_state = obs[:, :22] as f16, K padded to 32
    const int r = lane & 15, h = lane >> 4;
    const int b = bt * 16 + r;
    for (int j = 0; j < 16; ++j) {
      const int c = h * 16 + j;
      const float v = (c < 22) ? obs[b * 122 + c] : 0.f;
      as[r * 128 + c] = (_Float16)v;
    }
  }
  wave_lds_sync();
  wave_gemm_lds<8, 1>(as, 128, rw1p, cs, 128, lane, 0, 0);   // 16x32 @ 32x128
  wave_lds_sync();
  ln_relu_to_f16(cs, 128, as, 128, 128, lane, rb1, rg1, rbe1, 0);
  wave_lds_sync();
  wave_gemm_lds<8, 4>(as, 128, rw2p, cs, 128, lane, 0, 0);   // 16x128 @ 128x128
  wave_lds_sync();
  ln_relu_to_f16(cs, 128, as, 128, 128, lane, rb2, rg2, rbe2, 0);
  wave_lds_sync();
  {  // emit rf tile in A-fragment order so trunk kernel reads fragments directly
    const int m = lane & 15, kh = (lane >> 4) * 16;
#pragma unroll
    for (int ks = 0; ks < 4; ++ks) {
      v16h v = *(const v16h*)(as + m * 128 + ks * 32 + kh);
      *(v16h*)(rfp + (bt * 4 + ks) * 512 + lane * 16) = v;
    }
  }
}

// ---------------- kernel 2: footstep MLP + trunk MLP + heads + outputs ----------------
__global__ void __launch_bounds__(64) gait_trunk_kernel(
    const float* __restrict__ fop, const float* __restrict__ costs,
    const _Float16* __restrict__ wsH, const _Float16* __restrict__ rfp,
    const float* fb1, const float* fg1, const float* fbe1,
    const float* fb2, const float* fg2, const float* fbe2,
    const float* tb1, const float* tg1, const float* tbe1,
    const float* tb2, const float* tg2, const float* tbe2,
    const float* vw, const float* vb, const float* dw, const float* db,
    const float* na_emb, float* __restrict__ out) {
  __shared__ alignas(32) _Float16 aS[2][16 * 256];
  __shared__ alignas(32) float    cS[2][16 * 256];
  const int lane = threadIdx.x & 31, wave = threadIdx.x >> 5;
  const int task = blockIdx.x * 2 + wave;          // 512 b-tiles x 65 options
  const int bt = task & 511, o = task >> 9;
  _Float16* as = aS[wave];
  float*    cs = cS[wave];
  const _Float16* fw1p = wsH + OFF_FW1P;
  const _Float16* fw2p = wsH + OFF_FW2P;
  const _Float16* tw1p = wsH + OFF_TW1P;
  const _Float16* tw2p = wsH + OFF_TW2P;

  const int r = lane & 15, h = lane >> 4;
  const int b = bt * 16 + r;

  // warm trunk weights in L2 (global_prefetch_b8)
  __builtin_prefetch(tw1p + lane * 64, 0, 1);
  __builtin_prefetch(tw2p + lane * 64, 0, 1);

  // ---- per-row option features + mask (o uniform per wave; mask per-row via selects) ----
  float fo0 = -1.f, dx = 0.f, dy = 0.f, cost = 0.f;
  int mask = 1;
  if (o < NOPT) {  // wave-uniform branch
    const float cst = costs[b * NOPT + o];
    mask = (cst == __builtin_huge_valf());
    const float a0 = fop[(b * NOPT + o) * 3 + 0];
    const float a1 = fop[(b * NOPT + o) * 3 + 1];
    const float a2 = fop[(b * NOPT + o) * 3 + 2];
    fo0  = mask ? -1.f : a0;
    dx   = mask ? 0.f : a1;
    dy   = mask ? 0.f : a2;
    cost = mask ? 0.f : cst;
  }
  const int leg = (int)fminf(fmaxf(fo0, 0.f), 3.f);

  // fin = [one_hot(leg,4), dx, dy, safe_cost, 0-pad] -> a_stage cols 0..31 (f16)
  for (int j = 0; j < 16; ++j) {
    const int c = h * 16 + j;
    float v = 0.f;
    if (c < 4)       v = (!mask && c == leg) ? 1.f : 0.f;
    else if (c == 4) v = dx;
    else if (c == 5) v = dy;
    else if (c == 6) v = cost;
    as[r * 256 + c] = (_Float16)v;
  }
  // opts output (already masked / no-act substituted)
  if (h == 0) {
    const int base = (b * OPTS + o) * 3;
    out[base + 0] = fo0;
    out[base + 1] = dx;
    out[base + 2] = dy;
  }
  wave_lds_sync();

  // ---- footstep MLP: 7(->32) -> 64 -> 64 ----
  wave_gemm_lds<4, 1>(as, 256, fw1p, cs, 256, lane, 0, 0);
  wave_lds_sync();
  ln_relu_to_f16(cs, 256, as, 256, 64, lane, fb1, fg1, fbe1, 0);
  wave_lds_sync();
  wave_gemm_lds<4, 2>(as, 256, fw2p, cs, 256, lane, 0, 0);
  wave_lds_sync();
  {  // LN2 + na_emb select -> ff into a_stage cols 128..191
    const int c0 = h * 32, c1 = c0 + 32;
    float sum = 0.f, ss = 0.f;
    for (int c = c0; c < c1; ++c) {
      float x = cs[r * 256 + c] + fb2[c];
      sum += x; ss += x * x;
    }
    sum += __shfl_xor(sum, 16, 32);
    ss  += __shfl_xor(ss, 16, 32);
    const float mean = sum / 64.f;
    const float rstd = rsqrtf(ss / 64.f - mean * mean + 1e-5f);
    for (int c = c0; c < c1; ++c) {
      float x = cs[r * 256 + c] + fb2[c];
      float y = fmaxf((x - mean) * rstd * fg2[c] + fbe2[c], 0.f);
      float v = mask ? na_emb[c] : y;
      as[r * 256 + 128 + c] = (_Float16)v;
    }
  }
  wave_lds_sync();

  // ---- trunk MLP: [rf(128) | ff(64)] = 192 -> 256 -> 128 ----
  // two NT=8 halves to keep accumulator pressure at 64 VGPRs
  wave_gemm_trunk1<8>(as, 256, rfp + bt * 2048, tw1p, cs, 256, lane, 0);
  wave_gemm_trunk1<8>(as, 256, rfp + bt * 2048, tw1p + 8 * 6 * 512, cs, 256,
                      lane, 128);
  wave_lds_sync();
  ln_relu_to_f16(cs, 256, as, 256, 256, lane, tb1, tg1, tbe1, 0);
  wave_lds_sync();
  wave_gemm_lds<8, 8>(as, 256, tw2p, cs, 256, lane, 0, 0);
  wave_lds_sync();
  {  // final LN+ReLU kept in f32 (in place) for the heads
    const int c0 = h * 64, c1 = c0 + 64;
    float sum = 0.f, ss = 0.f;
    for (int c = c0; c < c1; ++c) {
      float x = cs[r * 256 + c] + tb2[c];
      sum += x; ss += x * x;
    }
    sum += __shfl_xor(sum, 16, 32);
    ss  += __shfl_xor(ss, 16, 32);
    const float mean = sum / 128.f;
    const float rstd = rsqrtf(ss / 128.f - mean * mean + 1e-5f);
    for (int c = c0; c < c1; ++c) {
      float x = cs[r * 256 + c] + tb2[c];
      cs[r * 256 + c] = fmaxf((x - mean) * rstd * tg2[c] + tbe2[c], 0.f);
    }
  }
  wave_lds_sync();
  {  // heads: value + duration (lane pair splits the 128-dot, combined via shfl)
    float vs = 0.f, ds = 0.f;
    const int c0 = h * 64;
    for (int c = c0; c < c0 + 64; ++c) {
      const float t = cs[r * 256 + c];
      vs += t * vw[c];
      ds += t * dw[c];
    }
    vs += __shfl_xor(vs, 16, 32);
    ds += __shfl_xor(ds, 16, 32);
    if (h == 0) {
      out[B_ * OPTS * 3 + b * OPTS + o] = vs + vb[0];
      const float sg = 1.f / (1.f + __expf(-(ds + db[0])));
      out[B_ * OPTS * 4 + b * OPTS + o] = mask ? 0.f : (0.1f + 0.2f * sg);
    }
  }
}

// ---------------- host launch ----------------
extern "C" void kernel_launch(void* const* d_in, const int* in_sizes, int n_in,
                              void* d_out, int out_size, void* d_ws, size_t ws_size,
                              hipStream_t stream) {
  (void)in_sizes; (void)n_in; (void)out_size; (void)ws_size;
  const float* obs   = (const float*)d_in[0];
  const float* fop   = (const float*)d_in[1];
  const float* costs = (const float*)d_in[2];
  const float* rw1 = (const float*)d_in[3];
  const float* rb1 = (const float*)d_in[4];
  const float* rg1 = (const float*)d_in[5];
  const float* rbe1= (const float*)d_in[6];
  const float* rw2 = (const float*)d_in[7];
  const float* rb2 = (const float*)d_in[8];
  const float* rg2 = (const float*)d_in[9];
  const float* rbe2= (const float*)d_in[10];
  const float* fw1 = (const float*)d_in[11];
  const float* fb1 = (const float*)d_in[12];
  const float* fg1 = (const float*)d_in[13];
  const float* fbe1= (const float*)d_in[14];
  const float* fw2 = (const float*)d_in[15];
  const float* fb2 = (const float*)d_in[16];
  const float* fg2 = (const float*)d_in[17];
  const float* fbe2= (const float*)d_in[18];
  const float* tw1 = (const float*)d_in[19];
  const float* tb1 = (const float*)d_in[20];
  const float* tg1 = (const float*)d_in[21];
  const float* tbe1= (const float*)d_in[22];
  const float* tw2 = (const float*)d_in[23];
  const float* tb2 = (const float*)d_in[24];
  const float* tg2 = (const float*)d_in[25];
  const float* tbe2= (const float*)d_in[26];
  const float* vw  = (const float*)d_in[27];
  const float* vb  = (const float*)d_in[28];
  const float* dw  = (const float*)d_in[29];
  const float* db  = (const float*)d_in[30];
  const float* na  = (const float*)d_in[31];

  _Float16* wsH = (_Float16*)d_ws;                 // needs ~2.3 MB workspace
  _Float16* rfp = wsH + OFF_RFP;
  float* out = (float*)d_out;

  gait_pack_kernel<<<128, 256, 0, stream>>>(rw1, rw2, fw1, fw2, tw1, tw2, wsH);
  gait_robot_kernel<<<128, 128, 0, stream>>>(obs, wsH, rfp,
                                             rb1, rg1, rbe1, rb2, rg2, rbe2);
  gait_trunk_kernel<<<16640, 64, 0, stream>>>(fop, costs, wsH, rfp,
                                              fb1, fg1, fbe1, fb2, fg2, fbe2,
                                              tb1, tg1, tbe1, tb2, tg2, tbe2,
                                              vw, vb, dw, db, na, out);
}
